// IDCT_IDXST_80599356277182
// MI455X (gfx1250) — matrix operand order repaired and checked by reading, not compiled, not run on hardware
//
#include <hip/hip_runtime.h>

// ---------------------------------------------------------------------------
// IDCT (rows) + IDXST (cols) as two split-precision f16 WMMA GEMMs.
//   Stage 0:  Y[b] = A_M^T * X[b]   A_M^T[i][k] = (k==0? 0.5 : cos(pi*k*(2i+1)/(2M)))
//   Stage 1:  Z[b] = Y[b]  * B_N    B_N[k][n]  = (k==0? 0.0 : sin(pi*k*(2n+1)/(2N)))
// Basis entries are generated on the fly with exact integer angle reduction
// (4N is a power of two), split into f16 hi+lo, and fed to
// v_wmma_f32_16x16x32_f16 with fp32 accumulation (3 products ~= fp32 GEMM).
// Register-level software pipelining hides global/trig latency under WMMA.
// ---------------------------------------------------------------------------

#define BM 128
#define BN 128
#define BK 32
#define LDA 40   // halves; 128x40x2B = 10KB per tile buffer, 4 buffers = 40KB LDS
#define LDB 40

typedef __attribute__((ext_vector_type(16))) _Float16 v16h;
typedef __attribute__((ext_vector_type(8)))  _Float16 v8h;
typedef __attribute__((ext_vector_type(2)))  _Float16 v2h;
typedef __attribute__((ext_vector_type(8)))  float    v8f;

__device__ __forceinline__ v16h join8(v8h a, v8h b) {
  return __builtin_shufflevector(a, b, 0,1,2,3,4,5,6,7,8,9,10,11,12,13,14,15);
}

// Split x,y into f16 hi/lo pairs and store packed (one ds_store_b32 each).
__device__ __forceinline__ void splitStore2(_Float16* hiP, _Float16* loP,
                                            float x, float y) {
  _Float16 hx = (_Float16)x, hy = (_Float16)y;
  v2h hi = { hx, hy };
  v2h lo = { (_Float16)(x - (float)hx), (_Float16)(y - (float)hy) };
  *(v2h*)hiP = hi;
  *(v2h*)loP = lo;
}

// STAGE 0: A = cos-basis (computed), B = X (global).  dst = Y (ws)
// STAGE 1: A = Y (global), B = sin-basis (computed).  dst = out
template<int STAGE>
__global__ __launch_bounds__(256)
void idct_gemm(const float* __restrict__ src, float* __restrict__ dst,
               int Mdim, int Ndim, int Kdim)
{
  __shared__ __align__(16) _Float16 sAhi[BM][LDA];
  __shared__ __align__(16) _Float16 sAlo[BM][LDA];
  __shared__ __align__(16) _Float16 sBhi[BN][LDB];
  __shared__ __align__(16) _Float16 sBlo[BN][LDB];

  const int tid    = threadIdx.x;
  const int lane   = tid & 31;
  const int l16    = lane & 15;
  const int hiHalf = lane >> 4;       // which half of the wave (K-split)
  const int wave   = tid >> 5;        // 8 waves
  const int wm     = wave >> 1;       // 0..3 : 32-row group
  const int wn     = wave & 1;        // 0..1 : 64-col group

  const int brow = blockIdx.y * BM;
  const int bcol = blockIdx.x * BN;
  const size_t plane = (size_t)Mdim * (size_t)Ndim;
  const float* gsrc = src + (size_t)blockIdx.z * plane;
  float*       gdst = dst + (size_t)blockIdx.z * plane;

  const float    ang   = 3.14159265358979323846f / (2.0f * (float)Kdim);
  const unsigned amask = (unsigned)(4 * Kdim - 1);   // 4*Kdim is a power of two

  // Each thread stages 8 K-adjacent pairs for A and for B (pair packs to b32).
  constexpr int APAIR = (BM * BK) / 512;   // 8
  constexpr int BPAIR = (BN * BK) / 512;   // 8
  float2 aReg[APAIR];
  float2 bReg[BPAIR];

  // --- prefetch tile k0 into registers (global loads or on-the-fly trig) ---
  auto prefetch = [&](int k0) {
    #pragma unroll
    for (int i = 0; i < APAIR; ++i) {
      int p = tid + i * 256;
      int e = p * 2;
      int r = e / BK, k = e % BK;              // k even; pair is (k, k+1), same row
      if (STAGE == 0) {
        int kk  = k0 + k;                       // even -> kk+1 never the k==0 slot
        int row = brow + r;
        int tw  = 2 * row + 1;
        aReg[i].x = (kk == 0) ? 0.5f
                  : __cosf(ang * (float)((unsigned)(kk * tw) & amask));
        aReg[i].y = __cosf(ang * (float)((unsigned)((kk + 1) * tw) & amask));
      } else {
        aReg[i] = *(const float2*)&gsrc[(size_t)(brow + r) * (size_t)Kdim +
                                        (size_t)(k0 + k)];
      }
    }
    #pragma unroll
    for (int i = 0; i < BPAIR; ++i) {
      int p = tid + i * 256;
      int c  = p % BN;                          // consecutive tids -> consecutive cols
      int k2 = (p / BN) * 2;                    // pair is (k2, k2+1), same col
      if (STAGE == 0) {
        const float* g = &gsrc[(size_t)(k0 + k2) * (size_t)Ndim + (size_t)(bcol + c)];
        bReg[i].x = g[0];
        bReg[i].y = g[(size_t)Ndim];
      } else {
        int kk  = k0 + k2;
        int col = bcol + c;
        int tw  = 2 * col + 1;
        bReg[i].x = (kk == 0) ? 0.0f
                  : __sinf(ang * (float)((unsigned)(kk * tw) & amask));
        bReg[i].y = __sinf(ang * (float)((unsigned)((kk + 1) * tw) & amask));
      }
    }
  };

  // --- split regs into hi/lo f16 and store to LDS (packed b32 stores) ---
  auto stageLDS = [&]() {
    #pragma unroll
    for (int i = 0; i < APAIR; ++i) {
      int p = tid + i * 256;
      int e = p * 2;
      int r = e / BK, k = e % BK;
      splitStore2(&sAhi[r][k], &sAlo[r][k], aReg[i].x, aReg[i].y);
    }
    #pragma unroll
    for (int i = 0; i < BPAIR; ++i) {
      int p = tid + i * 256;
      int c  = p % BN;
      int k2 = (p / BN) * 2;
      splitStore2(&sBhi[c][k2], &sBlo[c][k2], bReg[i].x, bReg[i].y);
    }
  };

  v8f acc[2][4] = {};

  prefetch(0);

  for (int k0 = 0; k0 < Kdim; k0 += BK) {
    stageLDS();                 // previous iter's end barrier makes this safe
    __syncthreads();

    if (k0 + BK < Kdim) prefetch(k0 + BK);   // overlaps with WMMA below

    // ---- A fragments (ISA 16-bit A 16x32 layout: lanes 0-15 K 0-7/16-23,
    //      lanes 16-31 K 8-15/24-31) ----
    v16h aHi[2], aLo[2];
    #pragma unroll
    for (int tm = 0; tm < 2; ++tm) {
      int row = wm * 32 + tm * 16 + l16;
      int kb  = hiHalf * 8;
      aHi[tm] = join8(*(const v8h*)&sAhi[row][kb], *(const v8h*)&sAhi[row][kb + 16]);
      aLo[tm] = join8(*(const v8h*)&sAlo[row][kb], *(const v8h*)&sAlo[row][kb + 16]);
    }
    // ---- B fragments (lanes 0-15 K 0-7/16-23, lanes 16-31 K 8-15/24-31; col = lane%16) ----
    #pragma unroll
    for (int tn = 0; tn < 4; ++tn) {
      int col = wn * 64 + tn * 16 + l16;
      int kb  = hiHalf * 8;
      v16h bHi = join8(*(const v8h*)&sBhi[col][kb], *(const v8h*)&sBhi[col][kb + 16]);
      v16h bLo = join8(*(const v8h*)&sBlo[col][kb], *(const v8h*)&sBlo[col][kb + 16]);
      #pragma unroll
      for (int tm = 0; tm < 2; ++tm) {
        acc[tm][tn] = __builtin_amdgcn_wmma_f32_16x16x32_f16(
            false, aHi[tm], false, bHi, (short)0, acc[tm][tn], false, false);
        acc[tm][tn] = __builtin_amdgcn_wmma_f32_16x16x32_f16(
            false, aLo[tm], false, bHi, (short)0, acc[tm][tn], false, false);
        acc[tm][tn] = __builtin_amdgcn_wmma_f32_16x16x32_f16(
            false, aHi[tm], false, bLo, (short)0, acc[tm][tn], false, false);
      }
    }
    __syncthreads();
  }

  // ---- epilogue: 16x16 f32 C layout (VGPR r -> row r / r+8 by lane half) ----
  #pragma unroll
  for (int tm = 0; tm < 2; ++tm) {
    int row0 = brow + wm * 32 + tm * 16 + hiHalf * 8;
    #pragma unroll
    for (int tn = 0; tn < 4; ++tn) {
      int col = bcol + wn * 64 + tn * 16 + l16;
      #pragma unroll
      for (int r = 0; r < 8; ++r) {
        gdst[(size_t)(row0 + r) * (size_t)Ndim + (size_t)col] = acc[tm][tn][r];
      }
    }
  }
}

extern "C" void kernel_launch(void* const* d_in, const int* in_sizes, int n_in,
                              void* d_out, int out_size, void* d_ws, size_t ws_size,
                              hipStream_t stream) {
  const float* X = (const float*)d_in[0];
  // expkM / expkN (d_in[1], d_in[2]) encode exp(i*pi*k/2K); we regenerate the
  // needed cos/sin values on the fly with exact integer angle reduction.
  const int Mdim = in_sizes[1] / 2;               // 4096
  const int Ndim = in_sizes[2] / 2;               // 4096
  const int Bsz  = in_sizes[0] / (Mdim * Ndim);   // 2

  float* Y   = (float*)d_ws;    // intermediate (B, M, N) fp32 — fully overwritten
  float* out = (float*)d_out;

  dim3 block(256);
  dim3 grid(Ndim / BN, Mdim / BM, Bsz);
  // Stage 0: IDCT along M  (Y = A_M^T * X), contraction length = M
  idct_gemm<0><<<grid, block, 0, stream>>>(X, Y, Mdim, Ndim, Mdim);
  // Stage 1: IDXST along N (Z = Y * B_N), contraction length = N
  idct_gemm<1><<<grid, block, 0, stream>>>(Y, out, Mdim, Ndim, Ndim);
}
